// MultiHeadAttention_34428457845229
// MI455X (gfx1250) — matrix operand order repaired and checked by reading, not compiled, not run on hardware
//
#include <hip/hip_runtime.h>

#define B_   2
#define S_   2048
#define D_   1024
#define H_   16
#define HD_  64
#define SCALE_ 0.125f   // 64^-0.5
#define AW   8          // waves per attention block

typedef _Float16 f16;
typedef __attribute__((ext_vector_type(16))) _Float16 v16h;
typedef __attribute__((ext_vector_type(8)))  _Float16 v8h;
typedef __attribute__((ext_vector_type(8)))  float    v8f;
typedef __attribute__((ext_vector_type(4)))  unsigned u32x4;
typedef __attribute__((ext_vector_type(8)))  int      i32x8;
typedef __attribute__((ext_vector_type(4)))  int      i32x4;

union FragA { v16h v; v8h h[2]; };

// A operand: matrix row-major [M x K], K contiguous. p pre-offset to (row0, k0).
__device__ __forceinline__ v16h load_a_frag(const f16* p, int ldk) {
  int lane = threadIdx.x & 31;
  int m = lane & 15, half = lane >> 4;
  const f16* base = p + (size_t)m * ldk + half * 8;
  FragA f;
  f.h[0] = *(const v8h*)(base);
  f.h[1] = *(const v8h*)(base + 16);
  return f.v;
}

// B operand: matrix [K x N] with N contiguous. Lane l holds column n = l%16; element s is k = 8h+s (s<8), 16+8h+(s-8) (s>=8), h = l/16.
__device__ __forceinline__ v16h load_b_frag(const f16* p, int ldn) {
  int lane = threadIdx.x & 31;
  int n = lane & 15, half = lane >> 4;
  v16h b;
  #pragma unroll
  for (int s = 0; s < 16; ++s) {
    int k = (s < 8) ? (8 * half + s) : (16 + 8 * half + (s - 8));
    b[s] = p[(size_t)k * ldn + n];
  }
  return b;
}

__device__ __forceinline__ v8f wmma16(v16h a, v16h b, v8f c) {
  return __builtin_amdgcn_wmma_f32_16x16x32_f16(false, a, false, b, (short)0, c, false, false);
}

// ---- Tensor Data Mover: 2-D tile load (global -> LDS), f16 elements ----
// D# per CDNA5 ISA. Groups 2/3 zero => 2-D (tile_dim2 = 0).
// This toolchain's builtin takes 6 args: (u32x4, i32x8, i32x4, i32x4, i32x8, cpol).
__device__ __forceinline__ void tdm_load_2d(unsigned lds_addr, const void* gaddr,
                                            unsigned tile0, unsigned tile1,
                                            unsigned stride0_elems) {
  unsigned long long ga = (unsigned long long)(uintptr_t)gaddr;
  u32x4 g0;
  g0.x = 1u;                                            // count=1 (valid), no gather
  g0.y = lds_addr;                                      // lds_addr[31:0]
  g0.z = (unsigned)ga;                                  // global_addr[31:0]
  g0.w = (unsigned)((ga >> 32) & 0x1FFFFFFu) | (2u << 30);  // addr[56:32], type=2
  i32x8 g1;
  g1[0] = (int)(1u << 16);                              // wg_mask=0, data_size=1 (2 bytes)
  g1[1] = (int)(tile0 << 16);                           // tensor_dim0[15:0] (== tile0)
  g1[2] = (int)((tile0 >> 16) | (tile1 << 16));         // tensor_dim0 hi | tensor_dim1 lo
  g1[3] = (int)((tile1 >> 16) | (tile0 << 16));         // tensor_dim1 hi | tile_dim0
  g1[4] = (int)tile1;                                   // tile_dim1 | tile_dim2=0
  g1[5] = (int)stride0_elems;                           // tensor_dim0_stride[31:0]
  g1[6] = 0;
  g1[7] = 0;
  i32x4 z4 = {};
  i32x8 z8 = {};
  __builtin_amdgcn_tensor_load_to_lds(g0, g1, z4, z4, z8, 0);
}

// ---------------- elementwise f32 -> f16 convert ----------------
__global__ void cvt_kernel(const float* __restrict__ src, f16* __restrict__ dst, int n) {
  int i = blockIdx.x * blockDim.x + threadIdx.x;
  if (i < n) dst[i] = (f16)src[i];
}

// ---------------- transpose + convert: Wt[k][n] = W[n][k], D_ x D_ ----------------
__global__ void cvt_t_kernel(const float* __restrict__ W, f16* __restrict__ Wt) {
  __shared__ float tile[32][33];
  int bx = blockIdx.x * 32;  // k block
  int by = blockIdx.y * 32;  // n block
  int tx = threadIdx.x, ty = threadIdx.y;  // 32 x 8
  #pragma unroll
  for (int i = 0; i < 32; i += 8)
    tile[ty + i][tx] = W[(size_t)(by + ty + i) * D_ + bx + tx];
  __syncthreads();
  #pragma unroll
  for (int i = 0; i < 32; i += 8)
    Wt[(size_t)(bx + ty + i) * D_ + by + tx] = (f16)tile[tx][ty + i];
}

// ---------------- GEMM: Y = X @ W^T + bias ----------------
// mode 0: f16 out, heads layout [B,H,S,HD]   (Q and V)
// mode 1: f16 out, transposed  [B,H,HD,S]    (K)
// mode 2: f32 out, row-major   [M x N]       (final projection)
__global__ void gemm_xwt_kernel(const f16* __restrict__ X, const f16* __restrict__ Wt,
                                const float* __restrict__ bias,
                                f16* __restrict__ Yh, float* __restrict__ Yf, int mode) {
  int wave = threadIdx.x >> 5, lane = threadIdx.x & 31;
  int m0 = blockIdx.x * 64 + wave * 16;
  int n0 = blockIdx.y * 64;

  v8f acc[4] = {};
  for (int k = 0; k < D_; k += 32) {
    v16h a = load_a_frag(X + (size_t)m0 * D_ + k, D_);
    #pragma unroll
    for (int c = 0; c < 4; ++c) {
      v16h b = load_b_frag(Wt + (size_t)k * D_ + n0 + c * 16, D_);
      acc[c] = wmma16(a, b, acc[c]);
    }
  }

  int half = lane >> 4, col = lane & 15;
  #pragma unroll
  for (int c = 0; c < 4; ++c) {
    int n = n0 + c * 16 + col;
    float bv = bias ? bias[n] : 0.0f;
    #pragma unroll
    for (int r = 0; r < 8; ++r) {
      int m = m0 + r + half * 8;
      float val = acc[c][r] + bv;
      if (mode == 2) {
        Yf[(size_t)m * D_ + n] = val;
      } else {
        int bb = m / S_, s = m % S_;
        int hh = n / HD_, d = n % HD_;
        if (mode == 0)
          Yh[(((size_t)bb * H_ + hh) * S_ + s) * HD_ + d] = (f16)val;
        else
          Yh[(((size_t)bb * H_ + hh) * HD_ + d) * S_ + s] = (f16)val;
      }
    }
  }
}

// ---------------- flash attention with TDM-staged K/V tiles ----------------
// Qh [B,H,S,HD]; Kt [B,H,HD,S]; Vh [B,H,S,HD]; AO f16 [B,S,D]
__global__ void attn_kernel(const f16* __restrict__ Qh, const f16* __restrict__ Kt,
                            const f16* __restrict__ Vh, f16* __restrict__ AO) {
  __shared__ f16 Pbuf[AW][16 * 32];
  __shared__ f16 Ktile[2][64 * 32];   // [d][key]  4KB each
  __shared__ f16 Vtile[2][32 * 64];   // [key][d]  4KB each

  int bh = blockIdx.x;
  int bb = bh / H_, hh = bh % H_;
  int wave = threadIdx.x >> 5, lane = threadIdx.x & 31;
  int half = lane >> 4, col = lane & 15;
  int m0 = blockIdx.y * (16 * AW) + wave * 16;   // query tile base

  const f16* Qp = Qh + ((size_t)bh * S_ + m0) * HD_;
  const f16* Kp = Kt + (size_t)bh * HD_ * S_;
  const f16* Vp = Vh + (size_t)bh * S_ * HD_;

  // Q fragments stay in registers for the whole key loop
  v16h qa0 = load_a_frag(Qp, HD_);        // k = 0..31 of head dim
  v16h qa1 = load_a_frag(Qp + 32, HD_);   // k = 32..63

  float mrun[8], lrun[8];
  #pragma unroll
  for (int r = 0; r < 8; ++r) { mrun[r] = -1e30f; lrun[r] = 0.0f; }
  v8f o[4] = {};

  f16* P = Pbuf[wave];

  // prologue: wave 0 kicks off DMA of the first K/V tiles into buffer 0
  if (wave == 0) {
    tdm_load_2d((unsigned)(uintptr_t)&Ktile[0][0], Kp,  32, 64, S_);
    tdm_load_2d((unsigned)(uintptr_t)&Vtile[0][0], Vp,  64, 32, HD_);
  }

  int cur = 0;
  for (int j = 0; j < S_; j += 32) {
    if (wave == 0) {
      if (j + 32 < S_) {
        int j2 = j + 32;
        // issue next tile's DMA, then release the current one (TENSORcnt is in-order)
        tdm_load_2d((unsigned)(uintptr_t)&Ktile[cur ^ 1][0], Kp + j2,               32, 64, S_);
        tdm_load_2d((unsigned)(uintptr_t)&Vtile[cur ^ 1][0], Vp + (size_t)j2 * HD_, 64, 32, HD_);
        __builtin_amdgcn_s_wait_tensorcnt(2);
      } else {
        __builtin_amdgcn_s_wait_tensorcnt(0);
      }
    }
    __syncthreads();   // current tile visible to all waves

    const f16* Kl = &Ktile[cur][0];
    const f16* Vl = &Vtile[cur][0];

    // scores: 16 rows x 32 keys, contraction over HD=64 (LDS B operands)
    v8f s0 = {}, s1 = {};
    s0 = wmma16(qa0, load_b_frag(Kl,           32), s0);
    s0 = wmma16(qa1, load_b_frag(Kl + 32 * 32, 32), s0);
    s1 = wmma16(qa0, load_b_frag(Kl + 16,           32), s1);
    s1 = wmma16(qa1, load_b_frag(Kl + 32 * 32 + 16, 32), s1);

    // online softmax on C-layout accumulators
    #pragma unroll
    for (int r = 0; r < 8; ++r) {
      float a  = s0[r] * SCALE_;
      float b2 = s1[r] * SCALE_;
      float bm = fmaxf(a, b2);
      #pragma unroll
      for (int msk = 1; msk < 16; msk <<= 1) bm = fmaxf(bm, __shfl_xor(bm, msk, 32));
      float mnew = fmaxf(mrun[r], bm);
      float corr = __expf(mrun[r] - mnew);
      float p0 = __expf(a - mnew);
      float p1 = __expf(b2 - mnew);
      float ps = p0 + p1;
      #pragma unroll
      for (int msk = 1; msk < 16; msk <<= 1) ps += __shfl_xor(ps, msk, 32);
      lrun[r] = lrun[r] * corr + ps;
      mrun[r] = mnew;
      o[0][r] *= corr; o[1][r] *= corr; o[2][r] *= corr; o[3][r] *= corr;
      int row = r + half * 8;
      P[row * 32 + col]      = (f16)p0;   // keys j..j+15
      P[row * 32 + 16 + col] = (f16)p1;   // keys j+16..j+31
    }

    // P (16x32) back as A fragment from this wave's private LDS tile
    v16h pa = load_a_frag(P, 32);
    #pragma unroll
    for (int c = 0; c < 4; ++c)
      o[c] = wmma16(pa, load_b_frag(Vl + c * 16, 64), o[c]);

    __syncthreads();   // everyone done reading buffer `cur` before it is re-filled
    cur ^= 1;
  }

  // normalize and write [B,S,D]
  #pragma unroll
  for (int r = 0; r < 8; ++r) {
    float inv = 1.0f / lrun[r];
    int s = m0 + r + half * 8;
    size_t base = ((size_t)bb * S_ + s) * D_ + hh * HD_;
    #pragma unroll
    for (int c = 0; c < 4; ++c)
      AO[base + c * 16 + col] = (f16)(o[c][r] * inv);
  }
}

extern "C" void kernel_launch(void* const* d_in, const int* in_sizes, int n_in,
                              void* d_out, int out_size, void* d_ws, size_t ws_size,
                              hipStream_t stream) {
  const float* q  = (const float*)d_in[0];
  const float* k  = (const float*)d_in[1];
  const float* v  = (const float*)d_in[2];
  const float* Wq = (const float*)d_in[3]; const float* bq = (const float*)d_in[4];
  const float* Wk = (const float*)d_in[5]; const float* bk = (const float*)d_in[6];
  const float* Wv = (const float*)d_in[7]; const float* bv = (const float*)d_in[8];
  const float* Wo = (const float*)d_in[9]; const float* bo = (const float*)d_in[10];
  float* out = (float*)d_out;

  const size_t NE_X = (size_t)B_ * S_ * D_;  // 4M elements
  const size_t NE_W = (size_t)D_ * D_;       // 1M elements
  f16* Xq  = (f16*)d_ws;
  f16* Xk  = Xq  + NE_X;
  f16* Xv  = Xk  + NE_X;
  f16* Wqt = Xv  + NE_X;
  f16* Wkt = Wqt + NE_W;
  f16* Wvt = Wkt + NE_W;
  f16* Wot = Wvt + NE_W;
  f16* Qh  = Wot + NE_W;
  f16* Kt  = Qh  + NE_X;
  f16* Vh  = Kt  + NE_X;
  f16* AO  = Vh  + NE_X;

  int ct = 256;
  int cb = (int)((NE_X + ct - 1) / ct);
  cvt_kernel<<<cb, ct, 0, stream>>>(q, Xq, (int)NE_X);
  cvt_kernel<<<cb, ct, 0, stream>>>(k, Xk, (int)NE_X);
  cvt_kernel<<<cb, ct, 0, stream>>>(v, Xv, (int)NE_X);

  dim3 tb(32, 8), tg(32, 32);
  cvt_t_kernel<<<tg, tb, 0, stream>>>(Wq, Wqt);
  cvt_t_kernel<<<tg, tb, 0, stream>>>(Wk, Wkt);
  cvt_t_kernel<<<tg, tb, 0, stream>>>(Wv, Wvt);
  cvt_t_kernel<<<tg, tb, 0, stream>>>(Wo, Wot);

  dim3 gg(64, 16);  // M/64 x N/64, 4 waves per block
  gemm_xwt_kernel<<<gg, 128, 0, stream>>>(Xq, Wqt, bq, Qh, nullptr, 0);
  gemm_xwt_kernel<<<gg, 128, 0, stream>>>(Xk, Wkt, bk, Kt, nullptr, 1);
  gemm_xwt_kernel<<<gg, 128, 0, stream>>>(Xv, Wvt, bv, Vh, nullptr, 0);

  attn_kernel<<<dim3(B_ * H_, S_ / (16 * AW)), 32 * AW, 0, stream>>>(Qh, Kt, Vh, AO);

  gemm_xwt_kernel<<<gg, 128, 0, stream>>>(AO, Wot, bo, nullptr, out, 2);
}
